// PhaseLM_14207751815390
// MI455X (gfx1250) — matrix operand order repaired and checked by reading, not compiled, not run on hardware
//
#include <hip/hip_runtime.h>

// ---------------- problem constants ----------------
#define B_   2
#define L_   4096
#define V_   8192
#define D_   256
#define KH_  256
#define H_   8
#define KT_  2048
#define NL_  4
#define HID_ 4096
#define M_   (B_*L_)        // 8192 rows everywhere
#define CH_  16             // scan chunks along L
#define LC_  (L_/CH_)       // 256

typedef __attribute__((ext_vector_type(16))) __bf16 v16bf;
typedef __attribute__((ext_vector_type(8)))  __bf16 bf16x8;
typedef __attribute__((ext_vector_type(8)))  float  v8f;

// ---- CDNA5 async global->LDS staging (ASYNCcnt), with sync fallback ----
#if defined(__AMDGCN__) && __has_builtin(__builtin_amdgcn_global_load_async_to_lds_b128) && \
    __has_builtin(__builtin_amdgcn_s_wait_asynccnt)
#define ASYNC_LDS 1
#else
#define ASYNC_LDS 0
#endif

#if ASYNC_LDS
typedef int av4i __attribute__((vector_size(16)));   // builtin's exact pointee type
#define GLD_ASYNC_B128(gp, lp)                                              \
  __builtin_amdgcn_global_load_async_to_lds_b128(                           \
      (__attribute__((address_space(1))) av4i*)(av4i*)(gp),                 \
      (__attribute__((address_space(3))) av4i*)(av4i*)(lp), 0, 0)
#define ASYNC_WAIT() __builtin_amdgcn_s_wait_asynccnt(0)
#else
#define ASYNC_WAIT()
#endif

__device__ __forceinline__ v16bf cat16(bf16x8 lo, bf16x8 hi) {
  return __builtin_shufflevector(lo, hi, 0,1,2,3,4,5,6,7,8,9,10,11,12,13,14,15);
}

// ---------------- fp32 -> bf16 weight staging ----------------
__global__ void k_cvt(const float* __restrict__ s, __bf16* __restrict__ d) {
  size_t i = (size_t)blockIdx.x * 256 + threadIdx.x;
  d[i] = (__bf16)s[i];
}

// ---------------- embedding gather -> bf16 ----------------
__global__ void k_embed(const int* __restrict__ tok, const float* __restrict__ emb,
                        __bf16* __restrict__ E) {
  int m = blockIdx.x, d = threadIdx.x;             // 8192 blocks x 256 threads
  E[(size_t)m * D_ + d] = (__bf16)emb[(size_t)tok[m] * D_ + d];
}

// ---------------- encoder: dual GEMM (z_r, z_i) + phase epilogue ----------------
// out[m,n] = atan2(zi,zr) * tanh(|z|+1e-12) * omega[n],  K = D_ = 256
template <bool B_BF16>
__global__ __launch_bounds__(256)
void k_encoder(const __bf16* __restrict__ E, const void* __restrict__ Wr,
               const void* __restrict__ Wi, const float* __restrict__ omega,
               float* __restrict__ out) {
  __shared__ __align__(16) __bf16 As[128][32];
  __shared__ __align__(16) __bf16 Brs[64][32];
  __shared__ __align__(16) __bf16 Bis[64][32];
  const int tid = threadIdx.x, lane = tid & 31, wave = tid >> 5;
  const int wm0 = (wave & 3) * 32, wn0 = (wave >> 2) * 32;
  const int bm = blockIdx.y * 128, bn = blockIdx.x * 64;
  v8f accR[2][2] = {}, accI[2][2] = {};

  for (int k0 = 0; k0 < D_; k0 += 32) {
    { // A tile 128x32 (bf16 in global)
      int row = tid >> 1, ks = (tid & 1) * 16;
      const __bf16* gp = &E[(size_t)(bm + row) * D_ + k0 + ks];
#if ASYNC_LDS
      GLD_ASYNC_B128(gp,     &As[row][ks]);
      GLD_ASYNC_B128(gp + 8, &As[row][ks + 8]);
#else
      *(bf16x8*)&As[row][ks]     = ((const bf16x8*)gp)[0];
      *(bf16x8*)&As[row][ks + 8] = ((const bf16x8*)gp)[1];
#endif
    }
    { // B tiles: Wr/Wi stored (N=2048, K=256) row-major => transposed access
      int n = tid >> 2, ks = (tid & 3) * 8;
      if (B_BF16) {
        const __bf16* gr = (const __bf16*)Wr + (size_t)(bn + n) * D_ + k0 + ks;
        const __bf16* gi = (const __bf16*)Wi + (size_t)(bn + n) * D_ + k0 + ks;
#if ASYNC_LDS
        GLD_ASYNC_B128(gr, &Brs[n][ks]);
        GLD_ASYNC_B128(gi, &Bis[n][ks]);
#else
        *(bf16x8*)&Brs[n][ks] = *(const bf16x8*)gr;
        *(bf16x8*)&Bis[n][ks] = *(const bf16x8*)gi;
#endif
      } else {
        const float* gr = (const float*)Wr + (size_t)(bn + n) * D_ + k0 + ks;
        const float* gi = (const float*)Wi + (size_t)(bn + n) * D_ + k0 + ks;
#pragma unroll
        for (int j = 0; j < 8; ++j) { Brs[n][ks + j] = (__bf16)gr[j]; Bis[n][ks + j] = (__bf16)gi[j]; }
      }
    }
    ASYNC_WAIT();
    __syncthreads();
    const int half = lane >> 4, lr = lane & 15;
    v16bf a[2], br[2], bi[2];
#pragma unroll
    for (int wm = 0; wm < 2; ++wm) {
      int ar = wm0 + wm * 16 + lr;
      a[wm] = cat16(*(const bf16x8*)&As[ar][half * 8], *(const bf16x8*)&As[ar][16 + half * 8]);
    }
#pragma unroll
    for (int wn = 0; wn < 2; ++wn) {
      int bc = wn0 + wn * 16 + lr;
      br[wn] = cat16(*(const bf16x8*)&Brs[bc][half * 8], *(const bf16x8*)&Brs[bc][16 + half * 8]);
      bi[wn] = cat16(*(const bf16x8*)&Bis[bc][half * 8], *(const bf16x8*)&Bis[bc][16 + half * 8]);
    }
#pragma unroll
    for (int wm = 0; wm < 2; ++wm)
#pragma unroll
      for (int wn = 0; wn < 2; ++wn) {
        accR[wm][wn] = __builtin_amdgcn_wmma_f32_16x16x32_bf16(false, a[wm], false, br[wn],
                                                               (short)0, accR[wm][wn], false, false);
        accI[wm][wn] = __builtin_amdgcn_wmma_f32_16x16x32_bf16(false, a[wm], false, bi[wn],
                                                               (short)0, accI[wm][wn], false, false);
      }
    __syncthreads();
  }
  const int lr = lane & 15, mofs = (lane >> 4) * 8;
#pragma unroll
  for (int wm = 0; wm < 2; ++wm)
#pragma unroll
    for (int wn = 0; wn < 2; ++wn)
#pragma unroll
      for (int j = 0; j < 8; ++j) {
        int row = bm + wm0 + wm * 16 + mofs + j;
        int col = bn + wn0 + wn * 16 + lr;
        float r = accR[wm][wn][j], im = accI[wm][wn][j];
        float mag = sqrtf(r * r + im * im) + 1e-12f;
        out[(size_t)row * KT_ + col] = atan2f(im, r) * tanhf(mag) * omega[col];
      }
}

// ---------------- generic WMMA GEMM, 128x64 block tile ----------------
enum { EPI_F32 = 0, EPI_BIAS_GELU_BF16 = 1, EPI_BIAS_RES_F32 = 2 };

template <int EPI, bool A_BF16, bool B_TRANS, bool B_BF16>
__global__ __launch_bounds__(256)
void k_gemm(const void* __restrict__ Ap, const void* __restrict__ Bp,
            const float* __restrict__ bias, const float* __restrict__ res,
            void* __restrict__ out, int Mtot, int Ntot, int Ktot) {
  __shared__ __align__(16) __bf16 As[128][32];
  __shared__ __align__(16) __bf16 Bs[64][32];          // stored [n][k]
  const int tid = threadIdx.x, lane = tid & 31, wave = tid >> 5;
  const int wm0 = (wave & 3) * 32, wn0 = (wave >> 2) * 32;
  const int bm = blockIdx.y * 128, bn = blockIdx.x * 64;
  v8f acc[2][2] = {};

  for (int k0 = 0; k0 < Ktot; k0 += 32) {
    { // A tile: each thread 16 contiguous K elements of one row
      int row = tid >> 1, ks = (tid & 1) * 16;
      if (A_BF16) {
        const __bf16* gp = (const __bf16*)Ap + (size_t)(bm + row) * Ktot + k0 + ks;
#if ASYNC_LDS
        GLD_ASYNC_B128(gp,     &As[row][ks]);
        GLD_ASYNC_B128(gp + 8, &As[row][ks + 8]);
#else
        *(bf16x8*)&As[row][ks]     = ((const bf16x8*)gp)[0];
        *(bf16x8*)&As[row][ks + 8] = ((const bf16x8*)gp)[1];
#endif
      } else {
        const float* gp = (const float*)Ap + (size_t)(bm + row) * Ktot + k0 + ks;
#pragma unroll
        for (int j = 0; j < 16; ++j) As[row][ks + j] = (__bf16)gp[j];
        if (k0 + 32 < Ktot) __builtin_prefetch(gp + 32, 0, 1);
      }
    }
    if (B_TRANS) { // global B is (N,K) row-major: contiguous K per row
      int n = tid >> 2, ks = (tid & 3) * 8;
      if (B_BF16) {
        const __bf16* g = (const __bf16*)Bp + (size_t)(bn + n) * Ktot + k0 + ks;
#if ASYNC_LDS
        GLD_ASYNC_B128(g, &Bs[n][ks]);
#else
        *(bf16x8*)&Bs[n][ks] = *(const bf16x8*)g;
#endif
      } else {
        const float* g = (const float*)Bp + (size_t)(bn + n) * Ktot + k0 + ks;
#pragma unroll
        for (int j = 0; j < 8; ++j) Bs[n][ks + j] = (__bf16)g[j];
      }
    } else {       // global B is (K,N) row-major: transpose into LDS
      int kk = tid & 31, ns = (tid >> 5) * 8;
      if (B_BF16) {
        const __bf16* g = (const __bf16*)Bp + (size_t)(k0 + kk) * Ntot + bn + ns;
        bf16x8 v = *(const bf16x8*)g;
#pragma unroll
        for (int j = 0; j < 8; ++j) Bs[ns + j][kk] = v[j];
        if (k0 + 32 < Ktot) __builtin_prefetch(g + (size_t)32 * Ntot, 0, 1);
      } else {
        const float* g = (const float*)Bp + (size_t)(k0 + kk) * Ntot + bn + ns;
#pragma unroll
        for (int j = 0; j < 8; ++j) Bs[ns + j][kk] = (__bf16)g[j];
        if (k0 + 32 < Ktot) __builtin_prefetch(g + (size_t)32 * Ntot, 0, 1);
      }
    }
    if (A_BF16 || (B_BF16 && B_TRANS)) ASYNC_WAIT();
    __syncthreads();
    const int half = lane >> 4, lr = lane & 15;
    v16bf a[2], b[2];
#pragma unroll
    for (int wm = 0; wm < 2; ++wm) {
      int ar = wm0 + wm * 16 + lr;
      a[wm] = cat16(*(const bf16x8*)&As[ar][half * 8], *(const bf16x8*)&As[ar][16 + half * 8]);
    }
#pragma unroll
    for (int wn = 0; wn < 2; ++wn) {
      int bc = wn0 + wn * 16 + lr;
      b[wn] = cat16(*(const bf16x8*)&Bs[bc][half * 8], *(const bf16x8*)&Bs[bc][16 + half * 8]);
    }
#pragma unroll
    for (int wm = 0; wm < 2; ++wm)
#pragma unroll
      for (int wn = 0; wn < 2; ++wn)
        acc[wm][wn] = __builtin_amdgcn_wmma_f32_16x16x32_bf16(false, a[wm], false, b[wn],
                                                              (short)0, acc[wm][wn], false, false);
    __syncthreads();
  }
  const int lr = lane & 15, mofs = (lane >> 4) * 8;
#pragma unroll
  for (int wm = 0; wm < 2; ++wm)
#pragma unroll
    for (int wn = 0; wn < 2; ++wn)
#pragma unroll
      for (int j = 0; j < 8; ++j) {
        int row = bm + wm0 + wm * 16 + mofs + j;
        int col = bn + wn0 + wn * 16 + lr;
        size_t idx = (size_t)row * Ntot + col;
        float c = acc[wm][wn][j];
        if (EPI == EPI_F32) {
          ((float*)out)[idx] = c;
        } else if (EPI == EPI_BIAS_GELU_BF16) {
          float xx = c + bias[col];
          ((__bf16*)out)[idx] = (__bf16)(0.5f * xx * (1.0f + erff(xx * 0.70710678118654752f)));
        } else {
          ((float*)out)[idx] = c + bias[col] + res[idx];
        }
      }
}

// ---------------- 3-pass causal scan + leaky mix ----------------
__global__ void k_scan1(const float* __restrict__ phi, float* __restrict__ part) {
  int t = blockIdx.x * 256 + threadIdx.x;
  int k = t % KT_, c = (t / KT_) % CH_, b = t / (KT_ * CH_);
  const float* p = phi + (size_t)b * L_ * KT_ + (size_t)c * LC_ * KT_ + k;
  float s = 0.f;
  for (int l = 0; l < LC_; ++l) s += p[(size_t)l * KT_];
  part[((size_t)b * CH_ + c) * KT_ + k] = s;
}
__global__ void k_scan2(float* __restrict__ part) {
  int t = blockIdx.x * 256 + threadIdx.x;         // over B_*KT_
  int k = t % KT_, b = t / KT_;
  float acc = 0.f;
  for (int c = 0; c < CH_; ++c) {
    size_t i = ((size_t)b * CH_ + c) * KT_ + k;
    float v = part[i]; part[i] = acc; acc += v;   // exclusive chunk offsets
  }
}
__global__ void k_scan3(const float* __restrict__ phi, const float* __restrict__ part,
                        const float* __restrict__ alphas, int layer, float* __restrict__ outM) {
  int t = blockIdx.x * 256 + threadIdx.x;
  int k = t % KT_, c = (t / KT_) % CH_, b = t / (KT_ * CH_);
  float a = 1.0f / (1.0f + expf(-alphas[layer]));
  const float* p = phi  + (size_t)b * L_ * KT_ + (size_t)c * LC_ * KT_ + k;
  float*       o = outM + (size_t)b * L_ * KT_ + (size_t)c * LC_ * KT_ + k;
  float acc = part[((size_t)b * CH_ + c) * KT_ + k];
  for (int l = 0; l < LC_; ++l) {
    float v = p[(size_t)l * KT_];
    acc += v;                                     // inclusive prefix
    o[(size_t)l * KT_] = v + a * (acc - v);
  }
}

// ---------------- phase norm: LN(cos), LN(sin), atan2, scale ----------------
template <bool ADD, bool WRITE_X>
__global__ __launch_bounds__(256)
void k_phasenorm(const float* __restrict__ in, float* __restrict__ phi,
                 const float* __restrict__ g, const float* __restrict__ bb,
                 __bf16* __restrict__ x) {
  __shared__ float red[4][256];
  const int row = blockIdx.x, tid = threadIdx.x;
  const float* ip = in + (size_t)row * KT_;
  float cv[8], sv[8];
  float sc = 0.f, sc2 = 0.f, ss = 0.f, ss2 = 0.f;
#pragma unroll
  for (int j = 0; j < 8; ++j) {
    float v = ip[tid + j * 256];
    cv[j] = cosf(v); sv[j] = sinf(v);
    sc += cv[j]; sc2 += cv[j] * cv[j]; ss += sv[j]; ss2 += sv[j] * sv[j];
  }
  red[0][tid] = sc; red[1][tid] = sc2; red[2][tid] = ss; red[3][tid] = ss2;
  __syncthreads();
  for (int off = 128; off > 0; off >>= 1) {
    if (tid < off) {
#pragma unroll
      for (int r = 0; r < 4; ++r) red[r][tid] += red[r][tid + off];
    }
    __syncthreads();
  }
  const float inv = 1.0f / (float)KT_;
  float mc = red[0][0] * inv, ms = red[2][0] * inv;
  float rc = rsqrtf(red[1][0] * inv - mc * mc + 1e-5f);
  float rs = rsqrtf(red[3][0] * inv - ms * ms + 1e-5f);
  float* pp = phi + (size_t)row * KT_;
#pragma unroll
  for (int j = 0; j < 8; ++j) {
    int c = tid + j * 256;
    float pn = g[c] * atan2f((sv[j] - ms) * rs, (cv[j] - mc) * rc) + bb[c];
    float ph = ADD ? (pp[c] + pn) : pn;
    pp[c] = ph;
    if (WRITE_X) {
      x[(size_t)row * (2 * KT_) + c]       = (__bf16)cosf(ph);
      x[(size_t)row * (2 * KT_) + KT_ + c] = (__bf16)sinf(ph);
    }
  }
}

// ---------------- launcher ----------------
extern "C" void kernel_launch(void* const* d_in, const int* in_sizes, int n_in,
                              void* d_out, int out_size, void* d_ws, size_t ws_size,
                              hipStream_t stream) {
  const int*   tokens = (const int*)  d_in[0];
  const float* emb    = (const float*)d_in[1];
  const float* W_real = (const float*)d_in[2];
  const float* W_imag = (const float*)d_in[3];
  const float* omega  = (const float*)d_in[4];
  const float* proj_W = (const float*)d_in[5];
  const float* alphas = (const float*)d_in[6];
  const float* ln_g   = (const float*)d_in[7];
  const float* ln_b   = (const float*)d_in[8];
  const float* w1     = (const float*)d_in[9];
  const float* b1     = (const float*)d_in[10];
  const float* w2     = (const float*)d_in[11];
  const float* b2     = (const float*)d_in[12];
  const float* fln_g  = (const float*)d_in[13];
  const float* fln_b  = (const float*)d_in[14];
  const float* fin_g  = (const float*)d_in[15];
  const float* fin_b  = (const float*)d_in[16];
  const float* head_W = (const float*)d_in[17];
  float* out = (float*)d_out;

  // activations workspace (~273 MB)
  char* ws = (char*)d_ws;  size_t o = 0;
  __bf16* E   = (__bf16*)(ws + o); o += (size_t)M_ * D_ * sizeof(__bf16);
  float*  phi = (float*) (ws + o); o += (size_t)M_ * KT_ * sizeof(float);
  float*  scr = (float*) (ws + o); o += (size_t)M_ * KT_ * sizeof(float);
  __bf16* x   = (__bf16*)(ws + o); o += (size_t)M_ * 2 * KT_ * sizeof(__bf16);
  __bf16* h   = (__bf16*)(ws + o); o += (size_t)M_ * HID_ * sizeof(__bf16);
  float* part = (float*) (ws + o); o += (size_t)B_ * CH_ * KT_ * sizeof(float);
  // bf16 weight staging (~245 MB): keeps each GEMM's re-read B panel L2-resident
  const size_t nWr = (size_t)H_ * KH_ * D_;            // 524288
  const size_t nPj = (size_t)KT_ * KT_;                // 4.19M
  const size_t nW1 = (size_t)NL_ * 2 * KT_ * HID_;     // 67.1M
  const size_t nW2 = (size_t)NL_ * HID_ * KT_;         // 33.6M
  const size_t nHd = (size_t)KT_ * V_;                 // 16.8M
  __bf16* wrB = (__bf16*)(ws + o); o += nWr * sizeof(__bf16);
  __bf16* wiB = (__bf16*)(ws + o); o += nWr * sizeof(__bf16);
  __bf16* pjB = (__bf16*)(ws + o); o += nPj * sizeof(__bf16);
  __bf16* w1B = (__bf16*)(ws + o); o += nW1 * sizeof(__bf16);
  __bf16* w2B = (__bf16*)(ws + o); o += nW2 * sizeof(__bf16);
  __bf16* hdB = (__bf16*)(ws + o); o += nHd * sizeof(__bf16);
  const bool wbf = (ws_size >= o);                     // bf16-weight path fits?

  if (wbf) {
    k_cvt<<<(int)(nWr / 256), 256, 0, stream>>>(W_real, wrB);
    k_cvt<<<(int)(nWr / 256), 256, 0, stream>>>(W_imag, wiB);
    k_cvt<<<(int)(nPj / 256), 256, 0, stream>>>(proj_W, pjB);
    k_cvt<<<(int)(nW1 / 256), 256, 0, stream>>>(w1, w1B);
    k_cvt<<<(int)(nW2 / 256), 256, 0, stream>>>(w2, w2B);
    k_cvt<<<(int)(nHd / 256), 256, 0, stream>>>(head_W, hdB);
  }

  k_embed<<<M_, 256, 0, stream>>>(tokens, emb, E);
  // encoder -> scr (raw phi), then proj (phi @ proj_W.T) -> phi
  dim3 gEnc(KT_ / 64, M_ / 128);
  if (wbf) k_encoder<true ><<<gEnc, 256, 0, stream>>>(E, wrB, wiB, omega, scr);
  else     k_encoder<false><<<gEnc, 256, 0, stream>>>(E, W_real, W_imag, omega, scr);
  if (wbf) k_gemm<EPI_F32, false, true, true ><<<gEnc, 256, 0, stream>>>(scr, pjB,    nullptr, nullptr, phi, M_, KT_, KT_);
  else     k_gemm<EPI_F32, false, true, false><<<gEnc, 256, 0, stream>>>(scr, proj_W, nullptr, nullptr, phi, M_, KT_, KT_);

  for (int i = 0; i < NL_; ++i) {
    k_scan1<<<(B_ * CH_ * KT_) / 256, 256, 0, stream>>>(phi, part);
    k_scan2<<<(B_ * KT_) / 256, 256, 0, stream>>>(part);
    k_scan3<<<(B_ * CH_ * KT_) / 256, 256, 0, stream>>>(phi, part, alphas, i, scr);
    // phi += phase_norm(mixed); emit x = [cos(phi), sin(phi)] bf16
    k_phasenorm<true, true><<<M_, 256, 0, stream>>>(scr, phi, ln_g + (size_t)i * KT_,
                                                    ln_b + (size_t)i * KT_, x);
    // h = gelu(x @ w1 + b1)
    dim3 g1(HID_ / 64, M_ / 128);
    if (wbf) k_gemm<EPI_BIAS_GELU_BF16, true, false, true ><<<g1, 256, 0, stream>>>(
        x, w1B + (size_t)i * 2 * KT_ * HID_, b1 + (size_t)i * HID_, nullptr, h, M_, HID_, 2 * KT_);
    else     k_gemm<EPI_BIAS_GELU_BF16, true, false, false><<<g1, 256, 0, stream>>>(
        x, w1  + (size_t)i * 2 * KT_ * HID_, b1 + (size_t)i * HID_, nullptr, h, M_, HID_, 2 * KT_);
    // scr = phi + h @ w2 + b2
    dim3 g2(KT_ / 64, M_ / 128);
    if (wbf) k_gemm<EPI_BIAS_RES_F32, true, false, true ><<<g2, 256, 0, stream>>>(
        h, w2B + (size_t)i * HID_ * KT_, b2 + (size_t)i * KT_, phi, scr, M_, KT_, HID_);
    else     k_gemm<EPI_BIAS_RES_F32, true, false, false><<<g2, 256, 0, stream>>>(
        h, w2  + (size_t)i * HID_ * KT_, b2 + (size_t)i * KT_, phi, scr, M_, KT_, HID_);
    // phi = phase_norm(scr)
    k_phasenorm<false, false><<<M_, 256, 0, stream>>>(scr, phi, fln_g + (size_t)i * KT_,
                                                      fln_b + (size_t)i * KT_, nullptr);
  }
  // final norm (in-place) + head GEMM -> logits fp32
  k_phasenorm<false, false><<<M_, 256, 0, stream>>>(phi, phi, fin_g, fin_b, nullptr);
  dim3 gH(V_ / 64, M_ / 128);
  if (wbf) k_gemm<EPI_F32, false, false, true ><<<gH, 256, 0, stream>>>(phi, hdB,    nullptr, nullptr, out, M_, V_, KT_);
  else     k_gemm<EPI_F32, false, false, false><<<gH, 256, 0, stream>>>(phi, head_W, nullptr, nullptr, out, M_, V_, KT_);
}
